// MultiHeadAttentionLayer_39857296507370
// MI455X (gfx1250) — compile-verified, matrix operand order checked by reading, not run on hardware
//
#include <hip/hip_runtime.h>
#include <hip/hip_bf16.h>

// Graph multi-head attention for MI455X (gfx1250, wave32, WMMA).
// Pipeline (6 kernels on `stream`):
//   1) zero edge-mask bitset (B*N*N bits = 512KB)
//   2) scatter edges + self-loops into bitset (atomicOr)
//   3) pack Wq/Wk/Wv/Wo once into transposed f16 (WT[n][k]) so every WMMA
//      B-fragment is two contiguous 16B loads per lane (global_load_b128)
//   4) Q/K/V projections (WMMA f16->f32); V written transposed [B][H][HD][N]
//   5) fused attention per (b,h,16-row q-tile): scores (1 WMMA per tile,
//      K=HD=32), masked softmax in f32 LDS, attn@V, deferred normalization
//   6) output projection -> f32 d_out
//
// Roofline: ~8.5 GFLOP total, ~45MB of traffic -> fully L2-resident (192MB).
// The win is keeping every fragment load vectorized (b128) and every GEMM on
// v_wmma_f32_16x16x32_f16.

typedef __attribute__((ext_vector_type(16))) _Float16 v16h;
typedef __attribute__((ext_vector_type(8)))  float    v8f;

#define BB     16
#define NN     512
#define DIN    256
#define DOUT   256
#define HH     8
#define HDIM   32
#define EE     131072
#define NODES  (BB * NN)              // 8192
#define MASKW  (BB * NN * (NN / 32))  // 131072 u32 words
#define LDH    (DIN + 8)              // padded LDS row (bank-conflict-free)
#define LDS_SC (NN + 4)               // padded score row

// ---------------------------------------------------------------------------
// WMMA fragment helpers (CDNA5 ISA §7.12.2, wave32):
//  A (16x32 f16): lanes 0-15 hold M=0-15 (half=0), lanes 16-31 same M (half=1)
//    element e of v16h holds K = 16*(e/8) + 8*half + (e%8)
//    -> per lane: two contiguous runs of 8 halves (16B each) => b128 loads
//  B (32x16 f16): symmetric layout on B^T, i.e. an A-style load on the
//    transposed matrix (lane = output column N, same K pattern)
//  C/D (16x16 f32): lane = N + 16*(M/8), VGPR v = M%8
// ---------------------------------------------------------------------------
__device__ __forceinline__ int kpat(int e, int half) {
    return ((e >> 3) << 4) + (half << 3) + (e & 7);
}

// A-style fragment from f16 row-major [row][ld]; also used for B-fragments by
// passing the transposed matrix with row0 = n0.
__device__ __forceinline__ v16h afrag_h(const _Float16* p, int ld, int row0, int k0) {
    int lane = threadIdx.x & 31;
    int m = lane & 15, half = lane >> 4;
    v16h a;
#pragma unroll
    for (int e = 0; e < 16; ++e)
        a[e] = p[(row0 + m) * ld + k0 + kpat(e, half)];
    return a;
}

// A fragment from f32 source (LDS scores), converted to f16
__device__ __forceinline__ v16h afrag_f32(const float* p, int ld, int row0, int k0) {
    int lane = threadIdx.x & 31;
    int m = lane & 15, half = lane >> 4;
    v16h a;
#pragma unroll
    for (int e = 0; e < 16; ++e)
        a[e] = (_Float16)p[(row0 + m) * ld + k0 + kpat(e, half)];
    return a;
}

// ---------------------------------------------------------------------------
// Kernel 1: zero the mask bitset
// ---------------------------------------------------------------------------
__global__ void k_zero_mask(unsigned* __restrict__ mask) {
    int i = blockIdx.x * blockDim.x + threadIdx.x;
    if (i < MASKW) mask[i] = 0u;
}

// ---------------------------------------------------------------------------
// Kernel 2: scatter edges + self-loops into bitset
// ---------------------------------------------------------------------------
__global__ void k_scatter_mask(const int* __restrict__ src,
                               const int* __restrict__ dst,
                               unsigned* __restrict__ mask) {
    int i = blockIdx.x * blockDim.x + threadIdx.x;
    int b, r, c;
    if (i < EE) {
        int s = src[i], d = dst[i];
        b = s / NN; r = s % NN; c = d % NN;
    } else if (i < EE + NODES) {      // self loops
        int g = i - EE;
        b = g >> 9; r = g & (NN - 1); c = r;
    } else {
        return;
    }
    atomicOr(&mask[(b * NN + r) * (NN / 32) + (c >> 5)], 1u << (c & 31));
}

// ---------------------------------------------------------------------------
// Kernel 3: pack weights -> transposed f16.  WT[n*K + k] = (f16)W[k*N + n]
// One pass over 4 x 64K elements (1MB read, 512KB written, once).
// ---------------------------------------------------------------------------
__global__ void k_pack_w(const float* __restrict__ Wq, const float* __restrict__ Wk,
                         const float* __restrict__ Wv, const float* __restrict__ Wo,
                         _Float16* __restrict__ WqT, _Float16* __restrict__ WkT,
                         _Float16* __restrict__ WvT, _Float16* __restrict__ WoT) {
    int i = blockIdx.x * blockDim.x + threadIdx.x;
    if (i >= DIN * DOUT) return;
    int n = i >> 8, k = i & 255;          // WT row-major [n][k]
    int j = k * DOUT + n;                 // W row-major [k][n]
    WqT[i] = (_Float16)Wq[j];
    WkT[i] = (_Float16)Wk[j];
    WvT[i] = (_Float16)Wv[j];
    WoT[i] = (_Float16)Wo[j];
}

// ---------------------------------------------------------------------------
// One 16x16 projection tile: acc = htile @ WT^T (+bias), K=256 in 8 WMMAs.
// VT=false: store [B][H][N][HD] (q,k).  VT=true: store [B][H][HD][N] (v),
// which makes each lane's 8 stores contiguous f16 AND makes the attention
// B-fragment for attn@V a pair of 16B loads.
// ---------------------------------------------------------------------------
template <bool VT>
__device__ __forceinline__ void proj_tile(const _Float16* ht, const _Float16* WT,
                                          const float* bias, _Float16* O,
                                          int mt, int nt) {
    v8f acc = {};
    for (int kt = 0; kt < DIN / 32; ++kt) {
        if (kt + 1 < DIN / 32)
            __builtin_prefetch(&WT[(size_t)(nt * 16) * DIN + kt * 32 + 32], 0, 0);
        v16h a   = afrag_h(ht, LDH, 0, kt * 32);
        v16h bfr = afrag_h(WT, DIN, nt * 16, kt * 32);   // B-frag on WT
        acc = __builtin_amdgcn_wmma_f32_16x16x32_f16(
            false, a, false, bfr, (short)0, acc, false, false);
    }
    const int lane = threadIdx.x & 31;
    const int n = lane & 15, mb = (lane >> 4) * 8;
    const int c = nt * 16 + n;
    const float bi = bias[c];
    const int head = c >> 5, d = c & 31;
#pragma unroll
    for (int v = 0; v < 8; ++v) {
        int gm = mt * 16 + mb + v;
        int bg = gm >> 9, node = gm & (NN - 1);   // 16-row tile never crosses graphs
        if (VT)
            O[((size_t)(bg * HH + head) * HDIM + d) * NN + node] = (_Float16)(acc[v] + bi);
        else
            O[((size_t)(bg * HH + head) * NN + node) * HDIM + d] = (_Float16)(acc[v] + bi);
    }
}

// ---------------------------------------------------------------------------
// Kernel 4: Q/K/V projections. 512 blocks (16-row M-tiles) x 512 threads
// (16 waves = 16 n-tiles); each wave runs q,k,v sequentially with uniform
// (SGPR) weight pointers -> global_load, no flat path.
// ---------------------------------------------------------------------------
__global__ __launch_bounds__(512) void k_proj_qkv(
    const float* __restrict__ h,
    const _Float16* __restrict__ WqT, const float* __restrict__ bq,
    const _Float16* __restrict__ WkT, const float* __restrict__ bk,
    const _Float16* __restrict__ WvT, const float* __restrict__ bv,
    _Float16* __restrict__ qb, _Float16* __restrict__ kb,
    _Float16* __restrict__ vtb) {
    __shared__ _Float16 ht[16 * LDH];   // padded: rows on distinct banks
    const int mt = blockIdx.x;
    const int tid = threadIdx.x;

    for (int i = tid; i < 16 * DIN; i += 512)
        ht[(i >> 8) * LDH + (i & 255)] =
            (_Float16)h[(size_t)(mt * 16 + (i >> 8)) * DIN + (i & 255)];
    __syncthreads();

    const int nt = tid >> 5;            // wave id == output n-tile
    proj_tile<false>(ht, WqT, bq, qb, mt, nt);
    proj_tile<false>(ht, WkT, bk, kb, mt, nt);
    proj_tile<true >(ht, WvT, bv, vtb, mt, nt);
}

// ---------------------------------------------------------------------------
// Kernel 5: fused attention for one (b, h, 16-row q-tile).
// grid = (32 qtiles, 8 heads, 16 graphs), 256 threads = 8 waves.
// ---------------------------------------------------------------------------
__global__ __launch_bounds__(256) void k_attn(
    const _Float16* __restrict__ qb, const _Float16* __restrict__ kb,
    const _Float16* __restrict__ vtb, const unsigned* __restrict__ mask,
    _Float16* __restrict__ att) {
    __shared__ float sc[16][LDS_SC];  // padded score / weight tile (~33KB)
    __shared__ float red[16][16];
    __shared__ float rowmax[16];
    __shared__ float rowinv[16];
    __shared__ float oacc[16][HDIM];

    const int qt = blockIdx.x, hh = blockIdx.y, b = blockIdx.z;
    const int tid = threadIdx.x, wave = tid >> 5, lane = tid & 31;
    const size_t base = (size_t)(b * HH + hh) * NN * HDIM;
    const _Float16* qh = qb + base;
    const _Float16* kh = kb + base;
    const _Float16* vt = vtb + base;   // [HD][N] within this (b,h)
    const float scale = 0.17677669529663687f;  // 1/sqrt(HD)

    // ---- Phase 1: scores = q @ k^T * scale (K=HD=32, one WMMA per tile) ----
    v16h qa = afrag_h(qh, HDIM, qt * 16, 0);
    for (int ct = wave * 4; ct < wave * 4 + 4; ++ct) {
        v16h kf = afrag_h(kh, HDIM, ct * 16, 0);   // B = k^T slice
        v8f acc = {};
        acc = __builtin_amdgcn_wmma_f32_16x16x32_f16(
            false, qa, false, kf, (short)0, acc, false, false);
        const int n = lane & 15, mb = (lane >> 4) * 8;
#pragma unroll
        for (int v = 0; v < 8; ++v)
            sc[mb + v][ct * 16 + n] = acc[v] * scale;
    }
    __syncthreads();

    // ---- Phase 2: masked softmax statistics (16 threads per row) ----
    const int r = tid >> 4, c0 = tid & 15;
    const int qrow = qt * 16 + r;
    const unsigned* mrow = mask + (size_t)(b * NN + qrow) * (NN / 32);
    float mx = -1e30f;
    for (int c = c0; c < NN; c += 16) {
        bool on = (mrow[c >> 5] >> (c & 31)) & 1u;
        float s = on ? sc[r][c] : -1e9f;
        sc[r][c] = s;
        mx = fmaxf(mx, s);
    }
    red[r][c0] = mx;
    __syncthreads();
    if (c0 == 0) {
        float m2 = red[r][0];
        for (int i = 1; i < 16; ++i) m2 = fmaxf(m2, red[r][i]);
        rowmax[r] = m2;
    }
    __syncthreads();
    const float rm = rowmax[r];
    float sum = 0.f;
    for (int c = c0; c < NN; c += 16) {
        float e = __expf(sc[r][c] - rm);
        sc[r][c] = e;            // unnormalized weights feed the WMMA
        sum += e;
    }
    red[r][c0] = sum;
    ((float*)oacc)[tid] = 0.f;   // zero 512-entry accumulator (2 per thread)
    ((float*)oacc)[tid + 256] = 0.f;
    __syncthreads();
    if (c0 == 0) {
        float s2 = 0.f;
        for (int i = 0; i < 16; ++i) s2 += red[r][i];
        rowinv[r] = 1.f / s2;
    }
    __syncthreads();

    // ---- Phase 3: out = w @ v ; 8 waves = 2 HD n-tiles x 4 K-groups ----
    const int nt = wave & 1, kg = wave >> 1;
    v8f acc = {};
    for (int kc = kg * 4; kc < kg * 4 + 4; ++kc) {
        v16h a   = afrag_f32(&sc[0][0], LDS_SC, 0, kc * 32);   // f32 LDS -> f16
        v16h bfr = afrag_h(vt, NN, nt * 16, kc * 32);          // B-frag on v^T
        acc = __builtin_amdgcn_wmma_f32_16x16x32_f16(
            false, a, false, bfr, (short)0, acc, false, false);
    }
    {
        const int n = lane & 15, mb = (lane >> 4) * 8;
#pragma unroll
        for (int v = 0; v < 8; ++v)
            atomicAdd(&oacc[mb + v][nt * 16 + n], acc[v]);  // ds_add_f32
    }
    __syncthreads();

    // ---- Write att in [B][N][H*HD] f16 with deferred 1/rowsum ----
#pragma unroll
    for (int i = 0; i < 2; ++i) {
        int idx = tid + i * 256;
        int rr = idx >> 5, d = idx & 31;
        float val = oacc[rr][d] * rowinv[rr];
        att[((size_t)(b * NN + qt * 16 + rr)) * DOUT + hh * HDIM + d] =
            (_Float16)val;
    }
}

// ---------------------------------------------------------------------------
// Kernel 6: output projection  out = att(8192x256,f16) @ Wo + bo  -> f32
// ---------------------------------------------------------------------------
__global__ __launch_bounds__(512) void k_out_proj(
    const _Float16* __restrict__ att, const _Float16* __restrict__ WoT,
    const float* __restrict__ bo, float* __restrict__ out) {
    __shared__ _Float16 at[16 * LDH];
    const int mt = blockIdx.x, tid = threadIdx.x;

    for (int i = tid; i < 16 * DOUT; i += 512)
        at[(i >> 8) * LDH + (i & 255)] =
            att[(size_t)(mt * 16 + (i >> 8)) * DOUT + (i & 255)];
    __syncthreads();

    const int nt = tid >> 5;             // wave id == n-tile
    v8f acc = {};
    for (int kt = 0; kt < DOUT / 32; ++kt) {
        if (kt + 1 < DOUT / 32)
            __builtin_prefetch(&WoT[(size_t)(nt * 16) * DOUT + kt * 32 + 32], 0, 0);
        v16h a   = afrag_h(at, LDH, 0, kt * 32);
        v16h bfr = afrag_h(WoT, DOUT, nt * 16, kt * 32);
        acc = __builtin_amdgcn_wmma_f32_16x16x32_f16(
            false, a, false, bfr, (short)0, acc, false, false);
    }
    const int lane = tid & 31, n = lane & 15, mb = (lane >> 4) * 8;
    const float bias = bo[nt * 16 + n];
#pragma unroll
    for (int v = 0; v < 8; ++v)
        out[(size_t)(mt * 16 + mb + v) * DOUT + nt * 16 + n] = acc[v] + bias;
}

// ---------------------------------------------------------------------------
extern "C" void kernel_launch(void* const* d_in, const int* in_sizes, int n_in,
                              void* d_out, int out_size, void* d_ws, size_t ws_size,
                              hipStream_t stream) {
    (void)in_sizes; (void)n_in; (void)out_size; (void)ws_size;
    const float* h  = (const float*)d_in[0];
    const int*   src = (const int*)d_in[1];
    const int*   dst = (const int*)d_in[2];
    const float* Wq = (const float*)d_in[3];
    const float* bq = (const float*)d_in[4];
    const float* Wk = (const float*)d_in[5];
    const float* bk = (const float*)d_in[6];
    const float* Wv = (const float*)d_in[7];
    const float* bv = (const float*)d_in[8];
    const float* Wo = (const float*)d_in[9];
    const float* bo = (const float*)d_in[10];
    float* out = (float*)d_out;

    // Workspace: mask (512KB) + 4 packed weights (512KB) + q/k/vT/att (16MB)
    char* ws = (char*)d_ws;
    unsigned* mask = (unsigned*)ws;  ws += (size_t)MASKW * 4;
    _Float16* WqT = (_Float16*)ws;   ws += (size_t)DIN * DOUT * 2;
    _Float16* WkT = (_Float16*)ws;   ws += (size_t)DIN * DOUT * 2;
    _Float16* WvT = (_Float16*)ws;   ws += (size_t)DIN * DOUT * 2;
    _Float16* WoT = (_Float16*)ws;   ws += (size_t)DIN * DOUT * 2;
    _Float16* qb  = (_Float16*)ws;   ws += (size_t)NODES * DOUT * 2;
    _Float16* kb  = (_Float16*)ws;   ws += (size_t)NODES * DOUT * 2;
    _Float16* vtb = (_Float16*)ws;   ws += (size_t)NODES * DOUT * 2;
    _Float16* att = (_Float16*)ws;

    k_zero_mask<<<(MASKW + 255) / 256, 256, 0, stream>>>(mask);
    k_scatter_mask<<<(EE + NODES + 255) / 256, 256, 0, stream>>>(src, dst, mask);
    k_pack_w<<<(DIN * DOUT + 255) / 256, 256, 0, stream>>>(Wq, Wk, Wv, Wo,
                                                           WqT, WkT, WvT, WoT);
    k_proj_qkv<<<NODES / 16, 512, 0, stream>>>(h, WqT, bq, WkT, bk, WvT, bv,
                                               qb, kb, vtb);
    k_attn<<<dim3(NN / 16, HH, BB), 256, 0, stream>>>(qb, kb, vtb, mask, att);
    k_out_proj<<<NODES / 16, 512, 0, stream>>>(att, WoT, bo, out);
}
